// FPN_71347996721752
// MI455X (gfx1250) — matrix operand order repaired and checked by reading, not compiled, not run on hardware
//
#include <hip/hip_runtime.h>
#include <hip/hip_bf16.h>
#include <math.h>

typedef __attribute__((ext_vector_type(16))) __bf16 v16bf;
typedef __attribute__((ext_vector_type(8)))  float  v8f;

#define BSZ  4
#define HH   32
#define WW   32
#define LL   1024      /* H*W */
#define NPOS 4096      /* B*L */
#define DIN_ 1024
#define DIMD 512
#define DIMM 896
#define DTR_ 32
#define XR   34        /* DTR + 2*DST */

#define LDS_STRIDE 40  /* bf16 elems per LDS row: 80B, 16B-aligned, 20-bank stride */

enum { ACT_NONE = 0, ACT_SOFTPLUS = 1, ACT_HSIG = 2 };

static __device__ __forceinline__ __bf16 f2bf(float f) {
  union { float f; unsigned u; } v; v.f = f;
  unsigned r = (v.u + 0x7fffu + ((v.u >> 16) & 1u)) >> 16;
  union { unsigned short s; __bf16 b; } o; o.s = (unsigned short)r;
  return o.b;
}
static __device__ __forceinline__ float bf2f(__bf16 b) {
  union { unsigned short s; __bf16 b; } i; i.b = b;
  union { float f; unsigned u; } v; v.u = ((unsigned)i.s) << 16;
  return v.f;
}

// ---------------------------------------------------------------------------
// Generic bf16 WMMA GEMM: out[n][m] = act(alpha[m] * sum_k A[n][k]*W[m][k] + beta[m])
// Block: 256 threads = 8 waves. Block tile 128 rows x 32 cols; each wave owns
// 16 rows x 32 cols via two v8f accumulators sharing one A fragment.
// Staging is fully 16B-vectorized: one global_load_b128 -> one ds_store_b128.
// Requires N % 128 == 0 and K % 32 == 0 (true at every call site).
// ---------------------------------------------------------------------------
__global__ __launch_bounds__(256)
void gemm_bf16(const __bf16* __restrict__ A, const __bf16* __restrict__ W,
               const float* __restrict__ alpha, const float* __restrict__ beta,
               float* __restrict__ outF, __bf16* __restrict__ outB,
               int N, int M, int K, int act)
{
  __shared__ __bf16 sA[128 * LDS_STRIDE];   // 128x32 tile, padded rows
  __shared__ __bf16 sW[32 * LDS_STRIDE];    // 32x32 tile, padded rows
  const int tid  = threadIdx.x;
  const int lane = tid & 31;
  const int wid  = tid >> 5;
  const int wrow = wid << 4;          // 0..112
  const int n0   = blockIdx.y * 128;
  const int m0   = blockIdx.x * 32;
  const int grp  = lane >> 4;
  const int l16  = lane & 15;

  v8f acc0 = {};
  v8f acc1 = {};

  for (int kk = 0; kk < K; kk += 32) {
    if (kk + 32 < K) {  // speculative prefetch of next tiles (global_prefetch_b8)
      __builtin_prefetch(A + (size_t)n0 * K + kk + 32, 0, 0);
      __builtin_prefetch(W + (size_t)m0 * K + kk + 32, 0, 0);
    }
    // A tile: 128x32 bf16 = 8KB; 256 threads x 2 x 16B vector transactions
#pragma unroll
    for (int i = 0; i < 2; ++i) {
      int idx = (tid + i * 256) * 8;          // bf16 element index in tile
      int r = idx >> 5, c = idx & 31;         // c in {0,8,16,24}
      uint4 av = *(const uint4*)(A + (size_t)(n0 + r) * K + kk + c);
      *(uint4*)(&sA[r * LDS_STRIDE + c]) = av;
    }
    // W tile: 32x32 bf16 = 2KB; 128 threads x 16B, zero-fill beyond M
    if (tid < 128) {
      int idx = tid * 8;
      int m = idx >> 5, c = idx & 31;
      uint4 wv = {0u, 0u, 0u, 0u};
      if (m0 + m < M) wv = *(const uint4*)(W + (size_t)(m0 + m) * K + kk + c);
      *(uint4*)(&sW[m * LDS_STRIDE + c]) = wv;
    }
    __syncthreads();

    v16bf af, bf0, bf1;
#pragma unroll
    for (int j = 0; j < 16; ++j) {
      // A 16x32 bf16 layout: lanes0-15 K in {0..7,16..23}, lanes16-31 K in {8..15,24..31}
      int ka = grp * 8 + (j & 7) + ((j & 8) ? 16 : 0);
      af[j] = sA[(wrow + l16) * LDS_STRIDE + ka];
      // B 32x16 bf16 layout: lanes0-15 K in {0..7,16..23}, lanes16-31 K in {8..15,24..31}, N=lane&15
      int kb = grp * 8 + (j & 7) + ((j & 8) ? 16 : 0);
      bf0[j] = sW[l16 * LDS_STRIDE + kb];
      bf1[j] = sW[(16 + l16) * LDS_STRIDE + kb];
    }
    acc0 = __builtin_amdgcn_wmma_f32_16x16x32_bf16(false, af, false, bf0,
                                                   (short)0, acc0, false, false);
    acc1 = __builtin_amdgcn_wmma_f32_16x16x32_bf16(false, af, false, bf1,
                                                   (short)0, acc1, false, false);
    __syncthreads();
  }

  const int ch0 = m0 + l16;
  const int ch1 = m0 + 16 + l16;
  float al0 = 1.0f, be0 = 0.0f, al1 = 1.0f, be1 = 0.0f;
  if (ch0 < M) { if (alpha) al0 = alpha[ch0]; if (beta) be0 = beta[ch0]; }
  if (ch1 < M) { if (alpha) al1 = alpha[ch1]; if (beta) be1 = beta[ch1]; }
#pragma unroll
  for (int i = 0; i < 8; ++i) {
    // D layout: row = i (+8 for lanes 16-31), col = lane&15
    int row = n0 + wrow + i + (grp << 3);
    float v0 = acc0[i] * al0 + be0;
    float v1 = acc1[i] * al1 + be1;
    if (act == ACT_SOFTPLUS) {
      v0 = (v0 > 20.0f) ? v0 : log1pf(__expf(v0));
      v1 = (v1 > 20.0f) ? v1 : log1pf(__expf(v1));
    } else if (act == ACT_HSIG) {
      v0 = fminf(fmaxf(v0 + 3.0f, 0.0f), 6.0f) * (1.0f / 6.0f);
      v1 = fminf(fmaxf(v1 + 3.0f, 0.0f), 6.0f) * (1.0f / 6.0f);
    }
    if (ch0 < M) {
      if (outF) outF[(size_t)row * M + ch0] = v0;
      if (outB) outB[(size_t)row * M + ch0] = f2bf(v0);
    }
    if (ch1 < M) {
      if (outF) outF[(size_t)row * M + ch1] = v1;
      if (outB) outB[(size_t)row * M + ch1] = f2bf(v1);
    }
  }
}

// ---------------------------------------------------------------------------
// Small helper kernels
// ---------------------------------------------------------------------------
__global__ void cvt_f32_bf16(const float* __restrict__ in, __bf16* __restrict__ out, int n) {
  int i = blockIdx.x * 256 + threadIdx.x;
  if (i < n) out[i] = f2bf(in[i]);
}

__global__ void bn_fold(const float* __restrict__ g, const float* __restrict__ b,
                        const float* __restrict__ m, const float* __restrict__ v,
                        const float* __restrict__ convb,
                        float* __restrict__ alpha, float* __restrict__ beta, int C) {
  int i = blockIdx.x * 256 + threadIdx.x;
  if (i >= C) return;
  float inv = g[i] * rsqrtf(v[i] + 1e-5f);
  alpha[i] = inv;
  beta[i]  = (convb[i] - m[i]) * inv + b[i];
}

// xcat[n][c] bf16: strided gather of x0(::4), x1(::2), x2 at 32x32
__global__ void pack_xcat(const float* __restrict__ x0, const float* __restrict__ x1,
                          const float* __restrict__ x2, __bf16* __restrict__ out) {
  size_t gid = (size_t)blockIdx.x * 256 + threadIdx.x;
  if (gid >= (size_t)NPOS * DIMM) return;
  int c = (int)(gid % DIMM);
  int n = (int)(gid / DIMM);
  int b = n / LL, hw = n % LL;
  int h = hw >> 5, w = hw & 31;
  float vv;
  if (c < 128)       vv = x0[(((size_t)b * 128 + c)        * 128 + h * 4) * 128 + w * 4];
  else if (c < 384)  vv = x1[(((size_t)b * 256 + (c - 128)) * 64 + h * 2) * 64 + w * 2];
  else               vv = x2[(((size_t)b * 512 + (c - 384)) * 32 + h)     * 32 + w];
  out[gid] = f2bf(vv);
}

// NCHW f32 -> [b*HW+hw][c] bf16
__global__ void pack_nchw_nc(const float* __restrict__ in, __bf16* __restrict__ out,
                             int C, int HW) {
  size_t total = (size_t)BSZ * C * HW;
  size_t gid = (size_t)blockIdx.x * 256 + threadIdx.x;
  if (gid >= total) return;
  int c = (int)(gid % C);
  size_t n = gid / C;
  int b = (int)(n / HW), hw = (int)(n % HW);
  out[gid] = f2bf(in[((size_t)b * C + c) * HW + hw]);
}

// slice columns [off,off+C) of yu[n][896] -> bf16 [n][C]
__global__ void pack_slice(const float* __restrict__ yu, __bf16* __restrict__ out,
                           int off, int C) {
  size_t total = (size_t)NPOS * C;
  size_t gid = (size_t)blockIdx.x * 256 + threadIdx.x;
  if (gid >= total) return;
  int c = (int)(gid % C);
  int n = (int)(gid / C);
  out[gid] = f2bf(yu[(size_t)n * DIMM + off + c]);
}

// first 32 cols of xdbl[n][34] -> bf16 [n][32]
__global__ void pack_dtsin(const float* __restrict__ xdbl, __bf16* __restrict__ out) {
  int i = blockIdx.x * 256 + threadIdx.x;
  if (i >= NPOS * DTR_) return;
  int n = i / DTR_, r = i % DTR_;
  out[i] = f2bf(xdbl[(size_t)n * XR + r]);
}

// depthwise 3x3 SAME conv + SiLU on xin[b][h][w][d], emits the 4 direction
// sequences xsl[k][(b*L + l)][d] in bf16 (reversals/transpose baked into index)
__global__ __launch_bounds__(256)
void dwconv_silu_pack(const float* __restrict__ xin, const float* __restrict__ cw,
                      __bf16* __restrict__ xsl0, __bf16* __restrict__ xsl1,
                      __bf16* __restrict__ xsl2, __bf16* __restrict__ xsl3) {
  size_t gid = (size_t)blockIdx.x * 256 + threadIdx.x;
  if (gid >= (size_t)BSZ * LL * DIN_) return;
  int d = (int)(gid % DIN_);
  int n = (int)(gid / DIN_);
  int hw = n % LL, b = n / LL;
  int h = hw >> 5, w = hw & 31;
  float s = 0.0f;
#pragma unroll
  for (int dy = -1; dy <= 1; ++dy) {
    int hh = h + dy;
    if (hh < 0 || hh >= HH) continue;
#pragma unroll
    for (int dx = -1; dx <= 1; ++dx) {
      int wn = w + dx;
      if (wn < 0 || wn >= WW) continue;
      float xv = xin[((size_t)(b * LL + hh * WW + wn)) * DIN_ + d];
      s += xv * cw[d * 9 + (dy + 1) * 3 + (dx + 1)];
    }
  }
  float sv = s / (1.0f + __expf(-s));   // SiLU
  __bf16 bv = f2bf(sv);
  int l0 = hw;              // row-major sequence
  int l1 = w * HH + h;      // column-major sequence
  size_t base = (size_t)b * LL;
  xsl0[(base + l0) * DIN_ + d] = bv;
  xsl1[(base + l1) * DIN_ + d] = bv;
  xsl2[(base + (LL - 1 - l0)) * DIN_ + d] = bv;
  xsl3[(base + (LL - 1 - l1)) * DIN_ + d] = bv;
}

// selective scan (DST=1): h_l = exp(dt*A)*h_{l-1} + dt*B_l*x_l ; y = h*C_l + D*x
// one thread per channel d, serial over L; direction remap folded into store.
__global__ __launch_bounds__(256)
void ss_scan(const float* __restrict__ dtl, const __bf16* __restrict__ xsl,
             const float* __restrict__ xdbl, const float* __restrict__ A_logs,
             const float* __restrict__ Ds, float* __restrict__ ym, int k) {
  int d = blockIdx.x * 256 + threadIdx.x;
  int b = blockIdx.y;
  float Ak = -__expf(A_logs[k * DIN_ + d]);
  float Dv = Ds[k * DIN_ + d];
  float h = 0.0f;
  size_t base = (size_t)b * LL;
  for (int l = 0; l < LL; ++l) {
    size_t r = base + l;
    float dt = dtl[r * DIN_ + d];
    float xs = bf2f(xsl[r * DIN_ + d]);
    float bm = xdbl[r * XR + 32];
    float cm = xdbl[r * XR + 33];
    float a = __expf(dt * Ak);
    h = a * h + dt * bm * xs;
    float y = h * cm + Dv * xs;
    int pos;
    if (k == 0)      pos = l;
    else if (k == 1) pos = ((l & 31) << 5) + (l >> 5);
    else if (k == 2) pos = LL - 1 - l;
    else { int l2 = LL - 1 - l; pos = ((l2 & 31) << 5) + (l2 >> 5); }
    ym[(base + pos) * DIN_ + d] += y;
  }
}

// layernorm over d=1024, affine, emit bf16
__global__ __launch_bounds__(256)
void layernorm_bf(const float* __restrict__ ym, const float* __restrict__ g,
                  const float* __restrict__ bta, __bf16* __restrict__ out) {
  __shared__ float red[256];
  int n = blockIdx.x, tid = threadIdx.x;
  float s = 0.f, s2 = 0.f;
  for (int d = tid; d < DIN_; d += 256) {
    float v = ym[(size_t)n * DIN_ + d];
    s += v; s2 += v * v;
  }
  red[tid] = s; __syncthreads();
  for (int st = 128; st > 0; st >>= 1) { if (tid < st) red[tid] += red[tid + st]; __syncthreads(); }
  float mu = red[0] / DIN_; __syncthreads();
  red[tid] = s2; __syncthreads();
  for (int st = 128; st > 0; st >>= 1) { if (tid < st) red[tid] += red[tid + st]; __syncthreads(); }
  float var = red[0] / DIN_ - mu * mu;
  float inv = rsqrtf(var + 1e-5f);
  for (int d = tid; d < DIN_; d += 256) {
    float v = (ym[(size_t)n * DIN_ + d] - mu) * inv * g[d] + bta[d];
    out[(size_t)n * DIN_ + d] = f2bf(v);
  }
}

// out = lf * bilerp(sig) + bilerp(gf), half-pixel centers, write NCHW
__global__ __launch_bounds__(256)
void fuse_gate(const float* __restrict__ lf, const float* __restrict__ gf,
               const float* __restrict__ sg, float* __restrict__ out,
               int C, int Hl, int Wl) {
  size_t total = (size_t)BSZ * C * Hl * Wl;
  size_t gid = (size_t)blockIdx.x * 256 + threadIdx.x;
  if (gid >= total) return;
  int w = (int)(gid % Wl);
  int h = (int)((gid / Wl) % Hl);
  int c = (int)((gid / ((size_t)Wl * Hl)) % C);
  int b = (int)(gid / ((size_t)Wl * Hl * C));
  float sc = 32.0f / (float)Hl;
  float fy = fminf(fmaxf(((float)h + 0.5f) * sc - 0.5f, 0.0f), 31.0f);
  float fx = fminf(fmaxf(((float)w + 0.5f) * sc - 0.5f, 0.0f), 31.0f);
  int y0 = (int)fy, x0 = (int)fx;
  int y1 = min(y0 + 1, 31), x1 = min(x0 + 1, 31);
  float ty = fy - (float)y0, tx = fx - (float)x0;
  size_t gb = (size_t)b * LL;
#define GIDX(yy, xx) ((gb + (size_t)(yy) * 32 + (xx)) * C + c)
  float g00 = gf[GIDX(y0, x0)], g01 = gf[GIDX(y0, x1)];
  float g10 = gf[GIDX(y1, x0)], g11 = gf[GIDX(y1, x1)];
  float s00 = sg[GIDX(y0, x0)], s01 = sg[GIDX(y0, x1)];
  float s10 = sg[GIDX(y1, x0)], s11 = sg[GIDX(y1, x1)];
#undef GIDX
  float gv = (g00 * (1.f - tx) + g01 * tx) * (1.f - ty) + (g10 * (1.f - tx) + g11 * tx) * ty;
  float sv = (s00 * (1.f - tx) + s01 * tx) * (1.f - ty) + (s10 * (1.f - tx) + s11 * tx) * ty;
  float lv = lf[((size_t)b * Hl * Wl + (size_t)h * Wl + w) * C + c];
  out[gid] = lv * sv + gv;
}

// ---------------------------------------------------------------------------
// Host orchestration
// ---------------------------------------------------------------------------
static inline int ceil_div(size_t a, size_t b) { return (int)((a + b - 1) / b); }

extern "C" void kernel_launch(void* const* d_in, const int* in_sizes, int n_in,
                              void* d_out, int out_size, void* d_ws, size_t ws_size,
                              hipStream_t stream) {
  (void)in_sizes; (void)n_in; (void)out_size; (void)ws_size;
  const float* x0 = (const float*)d_in[0];
  const float* x1 = (const float*)d_in[1];
  const float* x2 = (const float*)d_in[2];
  const float* down_w = (const float*)d_in[3];
  const float* down_b = (const float*)d_in[4];
  const float* up_w = (const float*)d_in[5];
  const float* up_b = (const float*)d_in[6];
  const float* W_in = (const float*)d_in[7];
  const float* conv_w = (const float*)d_in[8];
  const float* W_xproj = (const float*)d_in[9];
  const float* W_dtproj = (const float*)d_in[10];
  const float* b_dtproj = (const float*)d_in[11];
  const float* A_logs = (const float*)d_in[12];
  const float* Ds = (const float*)d_in[13];
  const float* ln_g = (const float*)d_in[14];
  const float* ln_b = (const float*)d_in[15];
  const float* W_out = (const float*)d_in[16];
  const int CH[3] = {128, 256, 512};
  const int HL[3] = {128, 64, 32};
  const size_t outOff[3] = {0, (size_t)4 * 128 * 128 * 128,
                               (size_t)4 * 128 * 128 * 128 + (size_t)4 * 256 * 64 * 64};

  char* ws = (char*)d_ws;
  size_t off = 0;
  auto alloc = [&](size_t bytes) { size_t o = off; off += (bytes + 255) & ~(size_t)255; return o; };

  // weights bf16
  size_t o_downw = alloc((size_t)DIMD * DIMM * 2);
  size_t o_win   = alloc((size_t)DIN_ * DIMD * 2);
  size_t o_wxp   = alloc((size_t)4 * XR * DIN_ * 2);
  size_t o_wdt   = alloc((size_t)4 * DIN_ * DTR_ * 2);
  size_t o_wout  = alloc((size_t)DIMD * DIN_ * 2);
  size_t o_upw   = alloc((size_t)DIMM * DIMD * 2);
  size_t o_gw[3][3], o_al[3][3], o_be[3][3];
  for (int i = 0; i < 3; ++i)
    for (int j = 0; j < 3; ++j) {
      o_gw[i][j] = alloc((size_t)CH[i] * CH[i] * 2);
      o_al[i][j] = alloc((size_t)CH[i] * 4);
      o_be[i][j] = alloc((size_t)CH[i] * 4);
    }
  // activations
  size_t o_xcat  = alloc((size_t)NPOS * DIMM * 2);             // reused as xg slices
  size_t o_xd    = alloc((size_t)NPOS * DIMD * 2);
  size_t o_xin   = alloc((size_t)NPOS * DIN_ * 4);             // reused as ym
  size_t o_xsl   = alloc((size_t)4 * NPOS * DIN_ * 2);         // reused as gf + sg
  size_t o_xdbl  = alloc((size_t)4 * NPOS * XR * 4);
  size_t o_dtsin = alloc((size_t)4 * NPOS * DTR_ * 2);
  size_t o_dtl   = alloc((size_t)4 * NPOS * DIN_ * 4);         // reused as lf
  size_t o_ymn   = alloc((size_t)NPOS * DIN_ * 2);
  size_t o_yss   = alloc((size_t)NPOS * DIMD * 2);
  size_t o_yu    = alloc((size_t)NPOS * DIMM * 4);
  size_t o_xl    = alloc(((size_t)65536 * 128 + 16384 * 256 + 4096 * 512) * 2);

  size_t o_ym = o_xin;      // xin dead after depthwise pack
  size_t o_xg = o_xcat;     // xcat dead after down GEMM
  size_t o_gf = o_xsl;      // xsl dead after scan
  size_t o_sg = o_xsl + (size_t)NPOS * DIMM * 4;
  size_t o_lf = o_dtl;      // dtl dead after scan

  auto BF = [&](size_t o) { return (__bf16*)(ws + o); };
  auto F  = [&](size_t o) { return (float*)(ws + o); };

  // --- weight conversions ---
  auto cvt = [&](const float* src, size_t dst, size_t n) {
    cvt_f32_bf16<<<ceil_div(n, 256), 256, 0, stream>>>(src, BF(dst), (int)n);
  };
  cvt(down_w, o_downw, (size_t)DIMD * DIMM);
  cvt(W_in, o_win, (size_t)DIN_ * DIMD);
  cvt(W_xproj, o_wxp, (size_t)4 * XR * DIN_);
  cvt(W_dtproj, o_wdt, (size_t)4 * DIN_ * DTR_);
  cvt(W_out, o_wout, (size_t)DIMD * DIN_);
  cvt(up_w, o_upw, (size_t)DIMM * DIMD);
  for (int i = 0; i < 3; ++i) {
    const int base = 17 + 18 * i;
    for (int j = 0; j < 3; ++j) {      // j: 0=le, 1=ge, 2=ga
      cvt((const float*)d_in[base + 2 * j], o_gw[i][j], (size_t)CH[i] * CH[i]);
      bn_fold<<<ceil_div(CH[i], 256), 256, 0, stream>>>(
          (const float*)d_in[base + 6 + 4 * j],      // bn{j+1}_g
          (const float*)d_in[base + 7 + 4 * j],      // bn{j+1}_b
          (const float*)d_in[base + 8 + 4 * j],      // bn{j+1}_m
          (const float*)d_in[base + 9 + 4 * j],      // bn{j+1}_v
          (const float*)d_in[base + 2 * j + 1],      // conv bias
          F(o_al[i][j]), F(o_be[i][j]), CH[i]);
    }
  }

  auto gemm = [&](size_t oA, size_t oW_, const float* alpha, const float* beta,
                  float* outF, __bf16* outB, int N, int M, int K, int act) {
    dim3 g((M + 31) / 32, N / 128);
    gemm_bf16<<<g, 256, 0, stream>>>(BF(oA), BF(oW_), alpha, beta, outF, outB, N, M, K, act);
  };

  // --- trunk ---
  pack_xcat<<<ceil_div((size_t)NPOS * DIMM, 256), 256, 0, stream>>>(x0, x1, x2, BF(o_xcat));
  gemm(o_xcat, o_downw, nullptr, down_b, nullptr, BF(o_xd), NPOS, DIMD, DIMM, ACT_NONE);
  gemm(o_xd, o_win, nullptr, nullptr, F(o_xin), nullptr, NPOS, DIN_, DIMD, ACT_NONE);
  dwconv_silu_pack<<<ceil_div((size_t)NPOS * DIN_, 256), 256, 0, stream>>>(
      F(o_xin), conv_w,
      BF(o_xsl), BF(o_xsl + (size_t)NPOS * DIN_ * 2),
      BF(o_xsl + (size_t)2 * NPOS * DIN_ * 2), BF(o_xsl + (size_t)3 * NPOS * DIN_ * 2));

  for (int k = 0; k < 4; ++k) {
    size_t xslk = o_xsl + (size_t)k * NPOS * DIN_ * 2;
    size_t xdblk = o_xdbl + (size_t)k * NPOS * XR * 4;
    size_t dtsk = o_dtsin + (size_t)k * NPOS * DTR_ * 2;
    size_t dtlk = o_dtl + (size_t)k * NPOS * DIN_ * 4;
    gemm(xslk, o_wxp + (size_t)k * XR * DIN_ * 2, nullptr, nullptr,
         F(xdblk), nullptr, NPOS, XR, DIN_, ACT_NONE);
    pack_dtsin<<<ceil_div((size_t)NPOS * DTR_, 256), 256, 0, stream>>>(F(xdblk), BF(dtsk));
    gemm(dtsk, o_wdt + (size_t)k * DIN_ * DTR_ * 2, nullptr, b_dtproj + k * DIN_,
         F(dtlk), nullptr, NPOS, DIN_, DTR_, ACT_SOFTPLUS);
  }

  hipMemsetAsync(F(o_ym), 0, (size_t)NPOS * DIN_ * 4, stream);
  for (int k = 0; k < 4; ++k) {
    ss_scan<<<dim3(DIN_ / 256, BSZ), 256, 0, stream>>>(
        F(o_dtl + (size_t)k * NPOS * DIN_ * 4),
        BF(o_xsl + (size_t)k * NPOS * DIN_ * 2),
        F(o_xdbl + (size_t)k * NPOS * XR * 4),
        A_logs, Ds, F(o_ym), k);
  }
  layernorm_bf<<<NPOS, 256, 0, stream>>>(F(o_ym), ln_g, ln_b, BF(o_ymn));
  gemm(o_ymn, o_wout, nullptr, nullptr, nullptr, BF(o_yss), NPOS, DIMD, DIN_, ACT_NONE);
  gemm(o_yss, o_upw, nullptr, up_b, F(o_yu), nullptr, NPOS, DIMM, DIMD, ACT_NONE);

  // --- gate fusion per scale ---
  const float* xl_src[3] = {x0, x1, x2};
  size_t xgOff = 0, xlOff = 0, gfOff = 0;
  int chOff = 0;
  for (int i = 0; i < 3; ++i) {
    int C = CH[i], Hl = HL[i];
    size_t Nl = (size_t)BSZ * Hl * Hl;
    size_t o_xgi = o_xg + xgOff;
    size_t o_xli = o_xl + xlOff;
    size_t o_lfi = o_lf + xlOff * 2;            // f32, same element count as xl bf16
    size_t o_gfi = o_gf + gfOff;
    size_t o_sgi = o_sg + gfOff;
    pack_slice<<<ceil_div((size_t)NPOS * C, 256), 256, 0, stream>>>(F(o_yu), BF(o_xgi), chOff, C);
    pack_nchw_nc<<<ceil_div(Nl * C, 256), 256, 0, stream>>>(xl_src[i], BF(o_xli), C, Hl * Hl);
    gemm(o_xli, o_gw[i][0], F(o_al[i][0]), F(o_be[i][0]), F(o_lfi), nullptr,
         (int)Nl, C, C, ACT_NONE);
    gemm(o_xgi, o_gw[i][1], F(o_al[i][1]), F(o_be[i][1]), F(o_gfi), nullptr,
         NPOS, C, C, ACT_NONE);
    gemm(o_xgi, o_gw[i][2], F(o_al[i][2]), F(o_be[i][2]), F(o_sgi), nullptr,
         NPOS, C, C, ACT_HSIG);
    fuse_gate<<<ceil_div(Nl * C, 256), 256, 0, stream>>>(
        F(o_lfi), F(o_gfi), F(o_sgi), (float*)d_out + outOff[i], C, Hl, Hl);
    xgOff += (size_t)NPOS * C * 2;
    xlOff += Nl * C * 2;
    gfOff += (size_t)NPOS * C * 4;
    chOff += C;
  }
}
